// SpikingPredictor_8658654068778
// MI455X (gfx1250) — matrix operand order repaired and checked by reading, not compiled, not run on hardware
//
#include <hip/hip_runtime.h>

typedef float    v8f  __attribute__((ext_vector_type(8)));
typedef _Float16 v8h  __attribute__((ext_vector_type(8)));
typedef _Float16 v16h __attribute__((ext_vector_type(16)));

#define B_     64
#define C_     128
#define T_     1024
#define KSZ    5
#define H_     512
#define ENC_   512
// ALPHA = exp(-1/20)
#define ALPHA_F 0.95122942450071400f
#define ONEMA_F 0.04877057549928600f

// half-precision LDS row pads: +8 halves => stride shifts 4 banks => conflict-free b128 reads
#define AH_PAD  (ENC_ + 8)     // 520 halves
#define MEM_PAD (H_ + 4)       // f32 mem rows
#define MH_PAD  (H_ + 8)       // f16 mem shadow rows
#define PH_PAD  (C_ + 8)       // 136 halves

#define CAT16(lo, hi) __builtin_shufflevector(lo, hi, 0,1,2,3,4,5,6,7,8,9,10,11,12,13,14,15)

// async copy: 16B per lane, INST_OFFSET added to BOTH lds and global addresses
#define ASYNC_LOAD16(J) \
  asm volatile("global_load_async_to_lds_b128 %0, %1, %2 offset:" #J \
               :: "v"(lds_dst), "v"(voff), "s"(gb) : "memory")

// ---------------------------------------------------------------------------
// Kernel 0: one-time weight preparation.
//   w1h = f16(w1) [n][k];  w2h = f16(w2) [n][k];
//   gth[n][k] = (k==n) ? 1 : og[n][k]   (identity + zero-diag mask folded in)
// ---------------------------------------------------------------------------
__global__ __launch_bounds__(256) void cvt_weights(
    const float* __restrict__ w1, const float* __restrict__ w2,
    const float* __restrict__ og, _Float16* __restrict__ w1h,
    _Float16* __restrict__ w2h,   _Float16* __restrict__ gth)
{
  const int idx = blockIdx.x * 256 + threadIdx.x;   // grid covers 344064 exactly
  if (idx < H_ * ENC_) {
    w1h[idx] = (_Float16)w1[idx];
  } else if (idx < H_ * ENC_ + C_ * H_) {
    const int j = idx - H_ * ENC_;
    w2h[j] = (_Float16)w2[j];
  } else {
    const int j = idx - (H_ * ENC_ + C_ * H_);
    const int n = j >> 7, k = j & (C_ - 1);
    gth[j] = (k == n) ? (_Float16)1.0f : (_Float16)og[j];
  }
}

// ---------------------------------------------------------------------------
// Kernel 1: fused depthwise-conv + GEMM.  One workgroup per timestep t.
//   enc tile -> f16 in LDS, then hid[t,b,:] = enc @ w1^T + b1 via
//   V_WMMA_F32_16X16X32_F16 (8 waves x 16 tiles x 16 k-steps).
// ---------------------------------------------------------------------------
__global__ __launch_bounds__(256) void fused_conv_gemm(
    const float* __restrict__ x, const float* __restrict__ cw,
    const float* __restrict__ cb, const _Float16* __restrict__ w1h,
    const float* __restrict__ b1, float* __restrict__ hid)
{
  __shared__ _Float16 enc_h[B_ * AH_PAD];   // 66.5 KB
  const int t   = blockIdx.x;
  const int tid = threadIdx.x;

  // ---- depthwise conv (f32 math, f16 store) ----
  for (int idx = tid; idx < B_ * ENC_; idx += 256) {
    const int b = idx >> 9;
    const int e = idx & (ENC_ - 1);
    const int c = e >> 2;
    float acc = cb[e];
    const float* xr = x + ((size_t)(b * C_ + c)) * T_;
#pragma unroll
    for (int k = 0; k < KSZ; ++k) {
      const int tt = t + k - 2;
      if (tt >= 0 && tt < T_) acc += xr[tt] * cw[e * KSZ + k];
    }
    enc_h[b * AH_PAD + e] = (_Float16)acc;
  }
  __syncthreads();

  const int lane  = tid & 31;
  const int wave  = tid >> 5;
  const int ln16  = lane & 15;
  const int koffA = (lane < 16) ? 0 : 16;   // f16 A frag: hi half-wave holds K=16..31
  const int koffB = (lane < 16) ? 0 : 16;   // f16 B frag: hi half-wave holds K=16..31
  const int rbase = (lane < 16) ? 0 : 8;    // f32 D frag: hi half-wave holds M+8

  for (int tile = wave; tile < 128; tile += 8) {
    const int m0 = (tile >> 5) * 16;
    const int n0 = (tile & 31) * 16;
    v8f acc = {};
    const _Float16* aBase = &enc_h[(m0 + ln16) * AH_PAD + koffA];
    const _Float16* bBase = &w1h[(size_t)(n0 + ln16) * ENC_ + koffB];
#pragma unroll 4
    for (int kc = 0; kc < ENC_; kc += 32) {
      v8h  alo = *(const v8h*)(aBase + kc);
      v8h  ahi = *(const v8h*)(aBase + kc + 8);
      v16h a   = CAT16(alo, ahi);
      v16h b   = *(const v16h*)(bBase + kc);
      acc = __builtin_amdgcn_wmma_f32_16x16x32_f16(false, a, false, b,
                                                   (short)0, acc, false, false);
    }
    const int   col  = n0 + ln16;
    const float bias = b1[col];
#pragma unroll
    for (int i = 0; i < 8; ++i) {
      const int row = m0 + rbase + i;
      hid[((size_t)t * B_ + row) * H_ + col] = acc[i] + bias;
    }
  }
}

// ---------------------------------------------------------------------------
// Kernel 2: sequential LIF scan.  4 workgroups x 16 batch rows; mem (f32)
// persistent in LDS with an f16 shadow feeding the WMMAs.  hid is streamed
// through a double-buffered LDS window via GLOBAL_LOAD_ASYNC_TO_LDS_B128,
// overlapping next-step HBM/L2 fetch with this step's serialized WMMA chain.
// Per step:
//   issue async copies for t+1 -> buf[(t+1)&1];  s_wait_asynccnt 8
//   P1 mem = alpha*mem + (1-alpha)*hid; spike>=1 -> reset (exact f32)
//   P2 preact = mem_h @ w2^T + b2      (16 f16 WMMAs / wave)
//   P3 out = preact_h @ (I+gate)^T     (4  f16 WMMAs / wave)
// ---------------------------------------------------------------------------
__global__ __launch_bounds__(256) void lif_scan(
    const float* __restrict__ hid, const _Float16* __restrict__ w2h,
    const float* __restrict__ b2,  const _Float16* __restrict__ gth,
    float* __restrict__ out)
{
  __shared__ float    mem_f[16 * MEM_PAD];   // 33 KB: exact recurrence state
  __shared__ _Float16 mem_h[16 * MH_PAD];    // 16.6 KB: WMMA A operand
  __shared__ _Float16 pre_h[16 * PH_PAD];    // 4.3 KB
  __shared__ float    hbuf[2 * 8192];        // 64 KB: hid double buffer
  const int tid = threadIdx.x;
  const int b0  = blockIdx.x * 16;

  for (int idx = tid; idx < 16 * MEM_PAD; idx += 256) mem_f[idx] = 0.0f;

  const unsigned hbuf_base = (unsigned)(size_t)&hbuf[0];  // LDS offset = addr[31:0]
  const unsigned voff      = (unsigned)tid * 16u;         // per-lane 16B chunk

  // prologue: stage hid for t=0 into buffer 0 (8 x 512B per wave)
  {
    const float*   gb      = hid + (size_t)b0 * H_;
    const unsigned lds_dst = hbuf_base + voff;
    ASYNC_LOAD16(0);     ASYNC_LOAD16(4096);  ASYNC_LOAD16(8192);  ASYNC_LOAD16(12288);
    ASYNC_LOAD16(16384); ASYNC_LOAD16(20480); ASYNC_LOAD16(24576); ASYNC_LOAD16(28672);
  }
  __syncthreads();

  const int lane  = tid & 31;
  const int wave  = tid >> 5;
  const int ln16  = lane & 15;
  const int koffA = (lane < 16) ? 0 : 16;
  const int koffB = (lane < 16) ? 0 : 16;
  const int rbase = (lane < 16) ? 0 : 8;

  for (int t = 0; t < T_; ++t) {
    // ---- issue async copies for step t+1 (clamped on final step) ----
    {
      const int      tn      = (t + 1 < T_) ? (t + 1) : t;
      const float*   gb      = hid + ((size_t)tn * B_ + b0) * H_;
      const unsigned lds_dst = hbuf_base + ((unsigned)((t + 1) & 1)) * 32768u + voff;
      ASYNC_LOAD16(0);     ASYNC_LOAD16(4096);  ASYNC_LOAD16(8192);  ASYNC_LOAD16(12288);
      ASYNC_LOAD16(16384); ASYNC_LOAD16(20480); ASYNC_LOAD16(24576); ASYNC_LOAD16(28672);
    }
    // in-order completion: <=8 outstanding means this step's 8 copies are done
    asm volatile("s_wait_asynccnt 0x8" ::: "memory");
    __syncthreads();

    const float* __restrict__ hcur = &hbuf[(t & 1) ? 8192 : 0];

    // ---- P1: LIF update (f32) + f16 shadow ----
#pragma unroll 4
    for (int i = 0; i < 32; ++i) {
      const int idx = tid + i * 256;           // 0..8191  == r*512 + e
      const int r = idx >> 9, e = idx & 511;
      float m = mem_f[r * MEM_PAD + e];
      const float h = hcur[idx];
      m = ALPHA_F * m + ONEMA_F * h;
      m = (m >= 1.0f) ? 0.0f : m;              // hard threshold + reset
      mem_f[r * MEM_PAD + e] = m;
      mem_h[r * MH_PAD + e]  = (_Float16)m;
    }
    __syncthreads();

    // ---- P2: preact = mem @ w2^T + b2  (8 waves, one 16x16 N-tile each) ----
    {
      const int n0 = wave * 16;
      v8f acc = {};
      const _Float16* aBase = &mem_h[ln16 * MH_PAD + koffA];
      const _Float16* bBase = &w2h[(size_t)(n0 + ln16) * H_ + koffB];
#pragma unroll 4
      for (int kc = 0; kc < H_; kc += 32) {
        v8h  alo = *(const v8h*)(aBase + kc);
        v8h  ahi = *(const v8h*)(aBase + kc + 8);
        v16h a   = CAT16(alo, ahi);
        v16h b   = *(const v16h*)(bBase + kc);
        acc = __builtin_amdgcn_wmma_f32_16x16x32_f16(false, a, false, b,
                                                     (short)0, acc, false, false);
      }
      const int   col  = n0 + ln16;
      const float bias = b2[col];
#pragma unroll
      for (int i = 0; i < 8; ++i)
        pre_h[(rbase + i) * PH_PAD + col] = (_Float16)(acc[i] + bias);
    }
    __syncthreads();

    // ---- P3: out = preact @ (I + gate)^T (identity folded into gth) ----
    {
      const int n0  = wave * 16;
      const int col = n0 + ln16;
      v8f acc = {};
      const _Float16* aBase = &pre_h[ln16 * PH_PAD + koffA];
      const _Float16* bBase = &gth[(size_t)col * C_ + koffB];
#pragma unroll
      for (int kc = 0; kc < C_; kc += 32) {
        v8h  alo = *(const v8h*)(aBase + kc);
        v8h  ahi = *(const v8h*)(aBase + kc + 8);
        v16h a   = CAT16(alo, ahi);
        v16h b   = *(const v16h*)(bBase + kc);
        acc = __builtin_amdgcn_wmma_f32_16x16x32_f16(false, a, false, b,
                                                     (short)0, acc, false, false);
      }
#pragma unroll
      for (int i = 0; i < 8; ++i) {
        const int row = rbase + i;
        out[((size_t)(b0 + row) * C_ + col) * T_ + t] = acc[i];
      }
    }
    __syncthreads();
  }
}

// ---------------------------------------------------------------------------
extern "C" void kernel_launch(void* const* d_in, const int* in_sizes, int n_in,
                              void* d_out, int out_size, void* d_ws, size_t ws_size,
                              hipStream_t stream) {
  const float* x  = (const float*)d_in[0];
  const float* cw = (const float*)d_in[1];
  const float* cb = (const float*)d_in[2];
  const float* w1 = (const float*)d_in[3];
  const float* b1 = (const float*)d_in[4];
  const float* w2 = (const float*)d_in[5];
  const float* b2 = (const float*)d_in[6];
  const float* og = (const float*)d_in[7];
  float* out = (float*)d_out;

  // workspace layout: hid (128 MiB f32) | w1h (512 KB) | w2h (128 KB) | gth (32 KB)
  float*    hid = (float*)d_ws;
  char*     p   = (char*)d_ws + (size_t)T_ * B_ * H_ * sizeof(float);
  _Float16* w1h = (_Float16*)p;                 p += (size_t)H_ * ENC_ * 2;
  _Float16* w2h = (_Float16*)p;                 p += (size_t)C_ * H_ * 2;
  _Float16* gth = (_Float16*)p;

  const int cvt_elems = H_ * ENC_ + C_ * H_ + C_ * C_;   // 344064
  cvt_weights<<<cvt_elems / 256, 256, 0, stream>>>(w1, w2, og, w1h, w2h, gth);
  fused_conv_gemm<<<T_, 256, 0, stream>>>(x, cw, cb, w1h, b1, hid);
  lif_scan<<<B_ / 16, 256, 0, stream>>>(hid, w2h, b2, gth, out);
}
